// FactorGraphLayer_75788992905484
// MI455X (gfx1250) — matrix operand order repaired and checked by reading, not compiled, not run on hardware
//
#include <hip/hip_runtime.h>

typedef __bf16 bf16_t;
typedef __attribute__((ext_vector_type(16))) __bf16 v16bf;
typedef __attribute__((ext_vector_type(8)))  __bf16 v8bf;
typedef __attribute__((ext_vector_type(8)))  float  v8f;

#define H   64
#define NCC 10
#define ECC 5

// ---------------------------------------------------------------------------
// WMMA helpers (gfx1250, wave32): D = A(16x32 bf16) x B(32x16 bf16) + C(f32)
// ---------------------------------------------------------------------------
__device__ __forceinline__ v8f wmma_bf16(v16bf a, v16bf b, v8f c) {
  return __builtin_amdgcn_wmma_f32_16x16x32_bf16(false, a, false, b,
                                                 (short)0, c, false, false);
}

// A-fragment: row-major source. rowptr = start of this lane's M row (M = lane&15).
// ISA 16-bit A layout: slot e -> K = (e&7) + 16*(e>>3) + 8*(lane>>4).
// => two runs of 8 consecutive bf16 (16B each).
template <typename P>
__device__ __forceinline__ v16bf load_A_frag(P rowptr, int k0, int half) {
  v8bf a = *(const v8bf*)(rowptr + k0 + 8 * half);
  v8bf b = *(const v8bf*)(rowptr + k0 + 16 + 8 * half);
  v16bf r;
#pragma unroll
  for (int i = 0; i < 8; ++i) { r[i] = a[i]; r[i + 8] = b[i]; }
  return r;
}

// B-fragment from row-major weight W[nOut][ldk] (y = x @ W^T => B[k][n] = W[n][k]).
// ISA 16-bit B layout: N = lane&15, slot e -> K = k0 + (e&7) + 16*(e>>3) + 8*(lane>>4)
// => two runs of 8 consecutive bf16 (16B each) per lane.
__device__ __forceinline__ v16bf load_B_frag(const bf16_t* __restrict__ W,
                                             int ldk, int n0, int k0, int lane) {
  const bf16_t* p = W + (size_t)(n0 + (lane & 15)) * ldk + k0 + 8 * (lane >> 4);
  v8bf a = *(const v8bf*)p;
  v8bf b = *(const v8bf*)(p + 16);
  v16bf r;
#pragma unroll
  for (int i = 0; i < 8; ++i) { r[i] = a[i]; r[i + 8] = b[i]; }
  return r;
}

// Dump a 16x64 f32 C-layout tile (4 n-tiles of v8f) to this wave's LDS slice as
// bf16 so it can be re-read in A layout. C layout: acc[nt][r] is
// (row m = r + 8*(lane>>4), col = nt*16 + (lane&15)). Same-wave LDS is in-order;
// the wait + memory clobber stops compiler/hw reordering before the A reload.
__device__ __forceinline__ void spill_tile_lds(bf16_t* myLds, const v8f (&acc)[4],
                                               int half, int llo) {
#pragma unroll
  for (int nt = 0; nt < 4; ++nt)
#pragma unroll
    for (int r = 0; r < 8; ++r)
      myLds[(r + 8 * half) * H + nt * 16 + llo] = (bf16_t)acc[nt][r];
  asm volatile("s_wait_dscnt 0" ::: "memory");
}

__device__ __forceinline__ float frelu(float x) { return x > 0.f ? x : 0.f; }

// edge_feats is stored TILE-MAJOR so each lane's 32 values for its 16-edge tile
// are contiguous (128B): offset(tile, lane, r, nt) = tile*1024 + lane*32 + r*4 + nt
// where the value at (r, nt) is edge row m = r + 8*(lane>>4), col = nt*16 + (lane&15).

// ---------------------------------------------------------------------------
// Utility kernels
// ---------------------------------------------------------------------------
__global__ void zero_kernel(float* __restrict__ p, size_t n) {
  size_t i = (size_t)blockIdx.x * blockDim.x + threadIdx.x;
  size_t stride = (size_t)gridDim.x * blockDim.x;
  for (; i < n; i += stride) p[i] = 0.f;
}

__global__ void f32_to_bf16_kernel(const float* __restrict__ s,
                                   bf16_t* __restrict__ d, int n) {
  int i = blockIdx.x * blockDim.x + threadIdx.x;
  int stride = gridDim.x * blockDim.x;
  for (; i < n; i += stride) d[i] = (bf16_t)s[i];
}

// node_feats = node_logits @ Wproj^T + bproj  (K = 10, scalar fp32)
__global__ void __launch_bounds__(256)
node_init_kernel(const float* __restrict__ nl, const float* __restrict__ Wp,
                 const float* __restrict__ bp, float* __restrict__ nf,
                 bf16_t* __restrict__ nfb, int N) {
  int t = blockIdx.x * blockDim.x + threadIdx.x;
  if (t >= N * H) return;
  int n = t >> 6, h = t & 63;
  float a = bp[h];
#pragma unroll
  for (int j = 0; j < NCC; ++j) a += nl[n * NCC + j] * Wp[h * NCC + j];
  nf[t] = a;
  nfb[t] = (bf16_t)a;
}

// edge_feats = relu(edge_logits @ We1^T + be1) @ We2^T + be2
// First layer computed scalar in fp32 directly in WMMA A layout; second layer WMMA.
// Output written in tile-major layout, vectorized.
__global__ void __launch_bounds__(256)
edge_init_kernel(const float* __restrict__ el, const float* __restrict__ We1,
                 const float* __restrict__ be1, const bf16_t* __restrict__ We2b,
                 const float* __restrict__ be2, float* __restrict__ efT, int E) {
  const int lane = threadIdx.x & 31;
  const int wave = threadIdx.x >> 5;
  const int half = lane >> 4, llo = lane & 15;
  const int tile = blockIdx.x * 8 + wave;
  const int ebase = tile * 16;
  if (ebase >= E) return;

  int myRow = ebase + llo;
  if (myRow >= E) myRow = E - 1;
  float el5[ECC];
#pragma unroll
  for (int j = 0; j < ECC; ++j) el5[j] = el[(size_t)myRow * ECC + j];

  v8f acc[4];
#pragma unroll
  for (int nt = 0; nt < 4; ++nt) {
    float bb = be2[nt * 16 + llo];
#pragma unroll
    for (int r = 0; r < 8; ++r) acc[nt][r] = bb;
  }
#pragma unroll
  for (int kt = 0; kt < 2; ++kt) {
    v16bf afrag;
#pragma unroll
    for (int i = 0; i < 16; ++i) {
      int k = kt * 32 + (i & 7) + ((i >> 3) << 4) + 8 * half;
      float t = be1[k];
#pragma unroll
      for (int j = 0; j < ECC; ++j) t += el5[j] * We1[k * ECC + j];
      afrag[i] = (bf16_t)frelu(t);
    }
#pragma unroll
    for (int nt = 0; nt < 4; ++nt)
      acc[nt] = wmma_bf16(afrag, load_B_frag(We2b, H, nt * 16, kt * 32, lane), acc[nt]);
  }
  float4* out = (float4*)(efT + (size_t)tile * (16 * H) + lane * 32);
#pragma unroll
  for (int r = 0; r < 8; ++r)
    out[r] = make_float4(acc[0][r], acc[1][r], acc[2][r], acc[3][r]);
}

// Fused per-iteration edge phase:
//   c   = [nf[src] | nf[dst]] @ Wattn^T + battn        (4 K-tiles)
//   msg = (c @ Wv^T + bv) @ Wo^T + bo                  (2+2 K-tiles via LDS relayout)
//   ef  = relu(ef + msg)                               (tile-major, b128 ld/st)
//   m   = relu(nf[src] + ef);  agg[dst] += m           (f32 atomics, L2-resident)
__global__ void __launch_bounds__(256)
edge_update_kernel(const float* __restrict__ nf, const bf16_t* __restrict__ nfb,
                   float* __restrict__ efT, float* __restrict__ agg,
                   const int* __restrict__ src, const int* __restrict__ dst,
                   const bf16_t* __restrict__ Wattnb, const bf16_t* __restrict__ Wvb,
                   const bf16_t* __restrict__ Wob, const float* __restrict__ battn,
                   const float* __restrict__ bv, const float* __restrict__ bo, int E) {
  __shared__ bf16_t lds[8 * 16 * H];
  const int lane = threadIdx.x & 31;
  const int wave = threadIdx.x >> 5;
  const int half = lane >> 4, llo = lane & 15;
  const int tile = blockIdx.x * 8 + wave;
  const int ebase = tile * 16;
  if (ebase >= E) return;
  bf16_t* myLds = lds + wave * (16 * H);

  // This wave's edge_feats chunk is one contiguous 128B region per lane:
  // prefetch it now (global_prefetch_b8); it's consumed after ~32 WMMAs.
  float* efLane = efT + (size_t)tile * (16 * H) + lane * 32;
  __builtin_prefetch(efLane, 0, 3);

  int myE = ebase + llo; if (myE >= E) myE = E - 1;
  int rowS = src[myE], rowD = dst[myE];

  // GEMM1: 16x128 @ 128x64
  v8f acc[4];
#pragma unroll
  for (int nt = 0; nt < 4; ++nt) {
    float bb = battn[nt * 16 + llo];
#pragma unroll
    for (int r = 0; r < 8; ++r) acc[nt][r] = bb;
  }
#pragma unroll
  for (int kt = 0; kt < 4; ++kt) {
    int idx = (kt < 2) ? rowS : rowD;
    v16bf afrag = load_A_frag(nfb + (size_t)idx * H, (kt & 1) * 32, half);
#pragma unroll
    for (int nt = 0; nt < 4; ++nt)
      acc[nt] = wmma_bf16(afrag, load_B_frag(Wattnb, 2 * H, nt * 16, kt * 32, lane), acc[nt]);
  }
  spill_tile_lds(myLds, acc, half, llo);

  // GEMM2: c @ Wv^T + bv
  v8f acc2[4];
#pragma unroll
  for (int nt = 0; nt < 4; ++nt) {
    float bb = bv[nt * 16 + llo];
#pragma unroll
    for (int r = 0; r < 8; ++r) acc2[nt][r] = bb;
  }
#pragma unroll
  for (int kt = 0; kt < 2; ++kt) {
    v16bf afrag = load_A_frag(myLds + llo * H, kt * 32, half);
#pragma unroll
    for (int nt = 0; nt < 4; ++nt)
      acc2[nt] = wmma_bf16(afrag, load_B_frag(Wvb, H, nt * 16, kt * 32, lane), acc2[nt]);
  }
  spill_tile_lds(myLds, acc2, half, llo);

  // GEMM3: @ Wo^T + bo
  v8f acc3[4];
#pragma unroll
  for (int nt = 0; nt < 4; ++nt) {
    float bb = bo[nt * 16 + llo];
#pragma unroll
    for (int r = 0; r < 8; ++r) acc3[nt][r] = bb;
  }
#pragma unroll
  for (int kt = 0; kt < 2; ++kt) {
    v16bf afrag = load_A_frag(myLds + llo * H, kt * 32, half);
#pragma unroll
    for (int nt = 0; nt < 4; ++nt)
      acc3[nt] = wmma_bf16(afrag, load_B_frag(Wob, H, nt * 16, kt * 32, lane), acc3[nt]);
  }

  // edge update (vectorized, tile-major) + message + scatter
  const float4* efv = (const float4*)efLane;
  float4* efw = (float4*)efLane;
#pragma unroll
  for (int r = 0; r < 8; ++r) {
    int e = ebase + r + 8 * half;
    float4 ev = efv[r];
    float v0 = frelu(ev.x + acc3[0][r]);   // GAMMA = 1.0
    float v1 = frelu(ev.y + acc3[1][r]);
    float v2 = frelu(ev.z + acc3[2][r]);
    float v3 = frelu(ev.w + acc3[3][r]);
    efw[r] = make_float4(v0, v1, v2, v3);
    if (e < E) {
      int si = src[e], di = dst[e];
      float vv[4] = {v0, v1, v2, v3};
#pragma unroll
      for (int nt = 0; nt < 4; ++nt) {
        int col = nt * 16 + llo;
        float mv = frelu(nf[(size_t)si * H + col] + vv[nt]);
        atomicAdd(&agg[(size_t)di * H + col], mv);
      }
    }
  }
}

// node_feats = relu(relu((nf+agg) @ Wg1^T + bg1) @ Wg2^T + bg2)
__global__ void __launch_bounds__(256)
node_update_kernel(float* __restrict__ nf, bf16_t* __restrict__ nfb,
                   const float* __restrict__ agg, const bf16_t* __restrict__ Wg1b,
                   const bf16_t* __restrict__ Wg2b, const float* __restrict__ bg1,
                   const float* __restrict__ bg2, int N) {
  __shared__ bf16_t lds[8 * 16 * H];
  const int lane = threadIdx.x & 31;
  const int wave = threadIdx.x >> 5;
  const int half = lane >> 4, llo = lane & 15;
  const int nbase = (blockIdx.x * 8 + wave) * 16;
  if (nbase >= N) return;
  bf16_t* myLds = lds + wave * (16 * H);

  // h = nf + agg  -> bf16 in LDS (A layout source)
#pragma unroll
  for (int r = 0; r < 8; ++r) {
    int n = nbase + r + 8 * half;
    int nc = n < N ? n : N - 1;
#pragma unroll
    for (int nt = 0; nt < 4; ++nt) {
      int col = nt * 16 + llo;
      size_t o = (size_t)nc * H + col;
      myLds[(r + 8 * half) * H + col] = (bf16_t)(nf[o] + agg[o]);
    }
  }
  asm volatile("s_wait_dscnt 0" ::: "memory");

  v8f a1[4];
#pragma unroll
  for (int nt = 0; nt < 4; ++nt) {
    float bb = bg1[nt * 16 + llo];
#pragma unroll
    for (int r = 0; r < 8; ++r) a1[nt][r] = bb;
  }
#pragma unroll
  for (int kt = 0; kt < 2; ++kt) {
    v16bf afrag = load_A_frag(myLds + llo * H, kt * 32, half);
#pragma unroll
    for (int nt = 0; nt < 4; ++nt)
      a1[nt] = wmma_bf16(afrag, load_B_frag(Wg1b, H, nt * 16, kt * 32, lane), a1[nt]);
  }
#pragma unroll
  for (int nt = 0; nt < 4; ++nt)
#pragma unroll
    for (int r = 0; r < 8; ++r) a1[nt][r] = frelu(a1[nt][r]);
  spill_tile_lds(myLds, a1, half, llo);

  v8f a2[4];
#pragma unroll
  for (int nt = 0; nt < 4; ++nt) {
    float bb = bg2[nt * 16 + llo];
#pragma unroll
    for (int r = 0; r < 8; ++r) a2[nt][r] = bb;
  }
#pragma unroll
  for (int kt = 0; kt < 2; ++kt) {
    v16bf afrag = load_A_frag(myLds + llo * H, kt * 32, half);
#pragma unroll
    for (int nt = 0; nt < 4; ++nt)
      a2[nt] = wmma_bf16(afrag, load_B_frag(Wg2b, H, nt * 16, kt * 32, lane), a2[nt]);
  }
#pragma unroll
  for (int r = 0; r < 8; ++r) {
    int n = nbase + r + 8 * half;
    if (n >= N) continue;
#pragma unroll
    for (int nt = 0; nt < 4; ++nt) {
      int col = nt * 16 + llo;
      float v = frelu(a2[nt][r]);
      nf[(size_t)n * H + col] = v;
      nfb[(size_t)n * H + col] = (bf16_t)v;
    }
  }
}

// Heads: one thread per row, float4 row loads (single pass over features).
__global__ void __launch_bounds__(256)
node_head_kernel(const float* __restrict__ nf, const float* __restrict__ Wnh,
                 const float* __restrict__ bnh, float* __restrict__ out, int N) {
  int n = blockIdx.x * blockDim.x + threadIdx.x;
  if (n >= N) return;
  const float4* row = (const float4*)(nf + (size_t)n * H);
  float a[NCC];
#pragma unroll
  for (int c = 0; c < NCC; ++c) a[c] = bnh[c];
#pragma unroll
  for (int j = 0; j < H / 4; ++j) {
    float4 v = row[j];
    float vv[4] = {v.x, v.y, v.z, v.w};
#pragma unroll
    for (int q = 0; q < 4; ++q)
#pragma unroll
      for (int c = 0; c < NCC; ++c) a[c] += vv[q] * Wnh[c * H + j * 4 + q];
  }
#pragma unroll
  for (int c = 0; c < NCC; ++c) out[(size_t)n * NCC + c] = a[c];
}

// edge_head reads the tile-major edge_feats: row e = (tile = e>>4, m = e&15),
// value (col = nt*16 + l) lives at tile*1024 + (l + 16*(m>>3))*32 + (m&7)*4 + nt.
__global__ void __launch_bounds__(256)
edge_head_kernel(const float* __restrict__ efT, const float* __restrict__ Weh,
                 const float* __restrict__ beh, float* __restrict__ out, int E) {
  int e = blockIdx.x * blockDim.x + threadIdx.x;
  if (e >= E) return;
  int tile = e >> 4, m = e & 15;
  int half = m >> 3, r = m & 7;
  const float4* base = (const float4*)(efT + (size_t)tile * (16 * H));
  float a[ECC];
#pragma unroll
  for (int c = 0; c < ECC; ++c) a[c] = beh[c];
#pragma unroll
  for (int l = 0; l < 16; ++l) {
    float4 v = base[(l + 16 * half) * 8 + r];
    float vv[4] = {v.x, v.y, v.z, v.w};
#pragma unroll
    for (int nt = 0; nt < 4; ++nt) {
      int col = nt * 16 + l;
#pragma unroll
      for (int c = 0; c < ECC; ++c) a[c] += vv[nt] * Weh[c * H + col];
    }
  }
#pragma unroll
  for (int c = 0; c < ECC; ++c) out[(size_t)e * ECC + c] = a[c];
}

// ---------------------------------------------------------------------------
extern "C" void kernel_launch(void* const* d_in, const int* in_sizes, int n_in,
                              void* d_out, int out_size, void* d_ws, size_t ws_size,
                              hipStream_t stream) {
  const float* node_logits = (const float*)d_in[0];
  const float* edge_logits = (const float*)d_in[1];
  const int*   edge_index  = (const int*)d_in[2];
  const float* Wproj = (const float*)d_in[3],  *bproj = (const float*)d_in[4];
  const float* We1   = (const float*)d_in[5],  *be1   = (const float*)d_in[6];
  const float* We2   = (const float*)d_in[7],  *be2   = (const float*)d_in[8];
  const float* Wg1   = (const float*)d_in[9],  *bg1   = (const float*)d_in[10];
  const float* Wg2   = (const float*)d_in[11], *bg2   = (const float*)d_in[12];
  const float* Wv    = (const float*)d_in[13], *bv    = (const float*)d_in[14];
  const float* Wo    = (const float*)d_in[15], *bo    = (const float*)d_in[16];
  const float* Wattn = (const float*)d_in[17], *battn = (const float*)d_in[18];
  const float* Wnh   = (const float*)d_in[19], *bnh   = (const float*)d_in[20];
  const float* Weh   = (const float*)d_in[21], *beh   = (const float*)d_in[22];

  const int N = in_sizes[0] / NCC;
  const int E = in_sizes[1] / ECC;
  const int* src = edge_index;
  const int* dst = edge_index + E;
  const int etiles = (E + 15) / 16;

  // workspace carve (256B aligned)
  char* ws = (char*)d_ws;
  size_t off = 0;
  auto carve = [&](size_t bytes) -> void* {
    off = (off + 255) & ~(size_t)255;
    void* p = ws + off;
    off += bytes;
    return p;
  };
  float*  nf     = (float*) carve((size_t)N * H * sizeof(float));
  bf16_t* nfb    = (bf16_t*)carve((size_t)N * H * sizeof(bf16_t));
  float*  agg    = (float*) carve((size_t)N * H * sizeof(float));
  float*  ef     = (float*) carve((size_t)etiles * 16 * H * sizeof(float));
  bf16_t* We2b   = (bf16_t*)carve(H * H * sizeof(bf16_t));
  bf16_t* Wattnb = (bf16_t*)carve(H * 2 * H * sizeof(bf16_t));
  bf16_t* Wvb    = (bf16_t*)carve(H * H * sizeof(bf16_t));
  bf16_t* Wob    = (bf16_t*)carve(H * H * sizeof(bf16_t));
  bf16_t* Wg1b   = (bf16_t*)carve(H * H * sizeof(bf16_t));
  bf16_t* Wg2b   = (bf16_t*)carve(H * H * sizeof(bf16_t));
  (void)ws_size; (void)n_in;

  const int T = 256;
  // bf16 weight conversion
  f32_to_bf16_kernel<<<16, T, 0, stream>>>(We2,   We2b,   H * H);
  f32_to_bf16_kernel<<<32, T, 0, stream>>>(Wattn, Wattnb, H * 2 * H);
  f32_to_bf16_kernel<<<16, T, 0, stream>>>(Wv,    Wvb,    H * H);
  f32_to_bf16_kernel<<<16, T, 0, stream>>>(Wo,    Wob,    H * H);
  f32_to_bf16_kernel<<<16, T, 0, stream>>>(Wg1,   Wg1b,   H * H);
  f32_to_bf16_kernel<<<16, T, 0, stream>>>(Wg2,   Wg2b,   H * H);

  node_init_kernel<<<(N * H + T - 1) / T, T, 0, stream>>>(node_logits, Wproj, bproj,
                                                          nf, nfb, N);
  const int edgeBlocks = (etiles + 7) / 8;
  edge_init_kernel<<<edgeBlocks, T, 0, stream>>>(edge_logits, We1, be1, We2b, be2, ef, E);

  const int nodeBlocks = (N + 16 * 8 - 1) / (16 * 8);
  for (int it = 0; it < 2; ++it) {
    zero_kernel<<<1024, T, 0, stream>>>(agg, (size_t)N * H);
    edge_update_kernel<<<edgeBlocks, T, 0, stream>>>(nf, nfb, ef, agg, src, dst,
                                                     Wattnb, Wvb, Wob, battn, bv, bo, E);
    node_update_kernel<<<nodeBlocks, T, 0, stream>>>(nf, nfb, agg, Wg1b, Wg2b,
                                                     bg1, bg2, N);
  }

  float* node_out = (float*)d_out;
  float* edge_out = node_out + (size_t)N * NCC;
  node_head_kernel<<<(N + T - 1) / T, T, 0, stream>>>(nf, Wnh, bnh, node_out, N);
  edge_head_kernel<<<(E + T - 1) / T, T, 0, stream>>>(ef, Weh, beh, edge_out, E);
}
